// GlobalLocalCrossAttention_90391881712433
// MI455X (gfx1250) — compile-verified, matrix operand order checked by reading, not run on hardware
//
#include <hip/hip_runtime.h>
#include <hip/hip_bf16.h>
#include <cstdint>
#include <cstddef>

// Problem constants
#define B_    64
#define N_    577
#define D_    768
#define H_    12
#define HD_   64
#define SEL_  58            // 1 CLS + 57 top-k
#define MTOT  (B_ * N_)     // 36928 flat token rows
#define MQ    (B_ * 64)     // 4096 (58 padded to 64)
#define VT_STRIDE 37120     // padded token stride for transposed V
#define LDST  72            // LDS row stride (64 + 8 pad) in bf16 elems

typedef __attribute__((ext_vector_type(4)))  __bf16 v4bf;
typedef __attribute__((ext_vector_type(8)))  __bf16 v8bf;
typedef __attribute__((ext_vector_type(16))) __bf16 v16bf;
typedef __attribute__((ext_vector_type(8)))  float  v8f;

__device__ __forceinline__ __bf16 f2bf(float f) {
  union { float f; uint32_t u; } x; x.f = f;
  uint32_t r = (x.u + 0x7FFFu + ((x.u >> 16) & 1u)) >> 16;
  uint16_t h = (uint16_t)r;
  return __builtin_bit_cast(__bf16, h);
}

__device__ __forceinline__ v8f zero8() {
  v8f z;
#pragma unroll
  for (int i = 0; i < 8; ++i) z[i] = 0.0f;
  return z;
}

// Build a 16x32 A-frag / 32x16 B-frag lane payload from a contiguous 32-elem
// K-run. Per ISA layout: lane-half g=0 takes K chunks [0..7] and [16..23],
// g=1 takes [8..15] and [24..31]. Two 16-byte loads.
__device__ __forceinline__ v16bf load_frag(const __bf16* p, int g) {
  v8bf c0 = *(const v8bf*)(p + (g << 3));
  v8bf c1 = *(const v8bf*)(p + 16 + (g << 3));
  v16bf r;
#pragma unroll
  for (int i = 0; i < 8; ++i) { r[i] = c0[i]; r[i + 8] = c1[i]; }
  return r;
}

__device__ __forceinline__ v8f wmma_bf16(v16bf a, v16bf b, v8f c) {
  return __builtin_amdgcn_wmma_f32_16x16x32_bf16(
      /*neg_a=*/false, a, /*neg_b=*/false, b,
      /*c_mod=*/(short)0, c, /*reuse_a=*/false, /*reuse_b=*/false);
}

// -------------------------------------------------------------------------
// Kernel 1: per-batch top-57 of accumulated_attention[b,0,1:577].
// gidx[b*64 + 0] = 0 (CLS), gidx[b*64 + 1..57] = top indices + 1, pad = 0.
// -------------------------------------------------------------------------
__global__ void topk_kernel(const float* __restrict__ acc, int* __restrict__ gidx) {
  const int b = blockIdx.x, tid = threadIdx.x;
  __shared__ float vals[576];
  __shared__ float rv[256];
  __shared__ int   ri[256];
  const float* row = acc + (size_t)b * N_ * N_ + 1;
  for (int i = tid; i < 576; i += 256) vals[i] = row[i];
  if (tid == 0) gidx[b * 64] = 0;
  if (tid >= SEL_ && tid < 64) gidx[b * 64 + tid] = 0;
  __syncthreads();
  for (int t = 0; t < SEL_ - 1; ++t) {
    float bv = -1e30f; int bi = 0x7fffffff;
    for (int i = tid; i < 576; i += 256) {
      float v = vals[i];
      if (v > bv || (v == bv && i < bi)) { bv = v; bi = i; }
    }
    rv[tid] = bv; ri[tid] = bi;
    __syncthreads();
    for (int s = 128; s > 0; s >>= 1) {
      if (tid < s) {
        float ov = rv[tid + s]; int oi = ri[tid + s];
        if (ov > rv[tid] || (ov == rv[tid] && oi < ri[tid])) { rv[tid] = ov; ri[tid] = oi; }
      }
      __syncthreads();
    }
    if (tid == 0) { gidx[b * 64 + 1 + t] = ri[0] + 1; vals[ri[0]] = -1e30f; }
    __syncthreads();
  }
}

// -------------------------------------------------------------------------
// Kernel 2: f32 -> bf16 conversion (vectorized 4/thread).
// -------------------------------------------------------------------------
__global__ void cvt_kernel(const float* __restrict__ src, __bf16* __restrict__ dst, int n) {
  int i = (blockIdx.x * 256 + threadIdx.x) * 4;
  if (i + 3 < n) {
    const float4 v = *(const float4*)(src + i);
    v4bf o;
    o[0] = f2bf(v.x); o[1] = f2bf(v.y); o[2] = f2bf(v.z); o[3] = f2bf(v.w);
    *(v4bf*)(dst + i) = o;
  } else {
    for (; i < n; ++i) dst[i] = f2bf(src[i]);
  }
}

// -------------------------------------------------------------------------
// Kernel 3: tiled bf16 WMMA GEMM  C = A @ W^T,  K = 768.
// A: Mrows x 768 bf16 row-major (optional row gather for Q path)
// W: 768 x 768 bf16 row-major (row = d_out), i.e. B-frag columns contiguous.
// mode 0: bf16 row-major out (ld 768)    [K projection, gathered Q]
// mode 1: bf16 transposed out, out[col*VT_STRIDE + row]  [V^T]
// mode 2: f32 row-major out + bias       [final projection]
// Block 128 threads = 4 waves, each wave a 64x64 tile of the 128x128 block.
// -------------------------------------------------------------------------
__global__ __launch_bounds__(128)
void gemm_kernel(const __bf16* __restrict__ A, const __bf16* __restrict__ W,
                 const int* __restrict__ gather, int Mrows, int mode,
                 __bf16* __restrict__ outb, float* __restrict__ outf,
                 const float* __restrict__ bias) {
  __shared__ __bf16 As[128 * LDST];
  __shared__ __bf16 Bs[128 * LDST];
  const int tid = threadIdx.x;
  const int lane = tid & 31, wid = tid >> 5;
  const int g = lane >> 4, ln = lane & 15;
  const int wM = wid >> 1, wN = wid & 1;
  const int m0 = blockIdx.x * 128, n0 = blockIdx.y * 128;

  v8f acc[4][4];
#pragma unroll
  for (int i = 0; i < 4; ++i)
#pragma unroll
    for (int j = 0; j < 4; ++j) acc[i][j] = zero8();

  const int gr = m0 + tid;
  const bool avalid = gr < Mrows;
  int arow = 0;
  if (avalid) arow = gather ? ((gr >> 6) * N_ + gather[gr]) : gr;
  const __bf16* ap = A + (size_t)arow * D_;
  const __bf16* bp = W + (size_t)(n0 + tid) * D_;
  const __bf16 bz = f2bf(0.0f);

  for (int kk = 0; kk < D_; kk += 32) {
#pragma unroll
    for (int j = 0; j < 4; ++j) {
      v8bf av;
      if (avalid) {
        av = *(const v8bf*)(ap + kk + j * 8);
      } else {
#pragma unroll
        for (int e = 0; e < 8; ++e) av[e] = bz;
      }
      *(v8bf*)&As[tid * LDST + j * 8] = av;
      *(v8bf*)&Bs[tid * LDST + j * 8] = *(const v8bf*)(bp + kk + j * 8);
    }
    __syncthreads();
    v16bf af[4], bfr[4];
#pragma unroll
    for (int mi = 0; mi < 4; ++mi)
      af[mi] = load_frag(&As[(wM * 64 + mi * 16 + ln) * LDST], g);
#pragma unroll
    for (int ni = 0; ni < 4; ++ni)
      bfr[ni] = load_frag(&Bs[(wN * 64 + ni * 16 + ln) * LDST], g);
#pragma unroll
    for (int mi = 0; mi < 4; ++mi)
#pragma unroll
      for (int ni = 0; ni < 4; ++ni)
        acc[mi][ni] = wmma_bf16(af[mi], bfr[ni], acc[mi][ni]);
    __syncthreads();
  }

#pragma unroll
  for (int mi = 0; mi < 4; ++mi) {
#pragma unroll
    for (int ni = 0; ni < 4; ++ni) {
      const int col = n0 + wN * 64 + ni * 16 + ln;
#pragma unroll
      for (int r = 0; r < 8; ++r) {
        const int row = m0 + wM * 64 + mi * 16 + g * 8 + r;
        if (row < Mrows) {
          if (mode == 0)      outb[(size_t)row * D_ + col] = f2bf(acc[mi][ni][r]);
          else if (mode == 1) outb[(size_t)col * VT_STRIDE + row] = f2bf(acc[mi][ni][r]);
          else                outf[(size_t)row * D_ + col] = acc[mi][ni][r] + bias[col];
        }
      }
    }
  }
}

// -------------------------------------------------------------------------
// Kernel 4: flash-style local attention per (batch, head).
// 4 waves; wave w owns 16 query rows. Online softmax with 16-lane shuffle
// reductions; P re-laid out via a small per-wave LDS tile for the P@V^T
// WMMAs. Output scattered (scaled by 1/l) into merged token buffer xb.
// -------------------------------------------------------------------------
__global__ __launch_bounds__(128)
void attn_kernel(const __bf16* __restrict__ qsel, const __bf16* __restrict__ kb,
                 const __bf16* __restrict__ vt, const int* __restrict__ gidx,
                 __bf16* __restrict__ xb) {
  const int b = blockIdx.x, h = blockIdx.y;
  const int tid = threadIdx.x, lane = tid & 31, w = tid >> 5;
  const int g = lane >> 4, ln = lane & 15;
  const float scale = 0.125f;  // HD^-0.5
  __shared__ __bf16 Plds[4][16][40];  // per-wave 16x32 P tile (+8 pad)

  const __bf16* qp = qsel + (size_t)(b * 64 + w * 16 + ln) * D_ + h * HD_;
  const v16bf aq0 = load_frag(qp, g);
  const v16bf aq1 = load_frag(qp + 32, g);

  v8f o[4];
  float mrow[8], lrow[8];
#pragma unroll
  for (int f = 0; f < 4; ++f) o[f] = zero8();
#pragma unroll
  for (int r = 0; r < 8; ++r) { mrow[r] = -1e30f; lrow[r] = 0.0f; }

  const __bf16* kbase = kb + (size_t)b * N_ * D_ + h * HD_;
  const __bf16* vbase = vt + (size_t)(h * HD_) * VT_STRIDE + (size_t)b * N_;

  for (int jt = 0; jt < 19; ++jt) {  // 19*32 = 608 >= 577 tokens
    const int tok0 = jt * 32;
    v8f sf[2];
#pragma unroll
    for (int half = 0; half < 2; ++half) {
      const int tcol = tok0 + half * 16 + ln;
      const int tcl = tcol < N_ ? tcol : (N_ - 1);
      const __bf16* kp = kbase + (size_t)tcl * D_;
      const v16bf bk0 = load_frag(kp, g);
      const v16bf bk1 = load_frag(kp + 32, g);
      v8f s = zero8();
      s = wmma_bf16(aq0, bk0, s);
      s = wmma_bf16(aq1, bk1, s);
      const bool ok = tcol < N_;
#pragma unroll
      for (int r = 0; r < 8; ++r) s[r] = ok ? s[r] * scale : -1e30f;
      sf[half] = s;
    }
#pragma unroll
    for (int r = 0; r < 8; ++r) {
      float t = fmaxf(sf[0][r], sf[1][r]);
#pragma unroll
      for (int msk = 8; msk >= 1; msk >>= 1) t = fmaxf(t, __shfl_xor(t, msk));
      const float mnew = fmaxf(mrow[r], t);
      const float corr = __expf(mrow[r] - mnew);
      const float p0 = __expf(sf[0][r] - mnew);
      const float p1 = __expf(sf[1][r] - mnew);
      float rs = p0 + p1;
#pragma unroll
      for (int msk = 8; msk >= 1; msk >>= 1) rs += __shfl_xor(rs, msk);
      lrow[r] = lrow[r] * corr + rs;
      mrow[r] = mnew;
#pragma unroll
      for (int f = 0; f < 4; ++f) o[f][r] *= corr;
      Plds[w][g * 8 + r][ln]      = f2bf(p0);
      Plds[w][g * 8 + r][16 + ln] = f2bf(p1);
    }
    const v16bf apf = load_frag(&Plds[w][ln][0], g);
#pragma unroll
    for (int f = 0; f < 4; ++f) {
      const __bf16* vp = vbase + (size_t)(f * 16 + ln) * VT_STRIDE + tok0;
      const v16bf bv = load_frag(vp, g);
      o[f] = wmma_bf16(apf, bv, o[f]);
    }
  }

#pragma unroll
  for (int r = 0; r < 8; ++r) {
    const int i = w * 16 + g * 8 + r;
    if (i < SEL_) {
      const float inv = 1.0f / lrow[r];
      const int tok = gidx[b * 64 + i];
      __bf16* dst = xb + (size_t)(b * N_ + tok) * D_ + h * HD_;
#pragma unroll
      for (int f = 0; f < 4; ++f) dst[f * 16 + ln] = f2bf(o[f][r] * inv);
    }
  }
}

// -------------------------------------------------------------------------
extern "C" void kernel_launch(void* const* d_in, const int* in_sizes, int n_in,
                              void* d_out, int out_size, void* d_ws, size_t ws_size,
                              hipStream_t stream) {
  const float* x   = (const float*)d_in[0];
  const float* acc = (const float*)d_in[1];
  const float* qw  = (const float*)d_in[2];
  const float* kw  = (const float*)d_in[3];
  const float* vw  = (const float*)d_in[4];
  const float* pw  = (const float*)d_in[5];
  const float* pb  = (const float*)d_in[6];
  float* out = (float*)d_out;

  char* p = (char*)d_ws;
  auto carve = [&](size_t bytes) -> void* {
    void* r = (void*)p;
    p += (bytes + 255) & ~(size_t)255;
    return r;
  };
  __bf16* xb   = (__bf16*)carve((size_t)MTOT * D_ * 2);       // bf16 x / merged tokens
  __bf16* kbb  = (__bf16*)carve((size_t)MTOT * D_ * 2);       // K projection
  __bf16* vtb  = (__bf16*)carve((size_t)D_ * VT_STRIDE * 2);  // V^T (d, token) padded
  __bf16* qsel = (__bf16*)carve((size_t)MQ * D_ * 2);         // gathered Q (58->64 pad)
  __bf16* wqb  = (__bf16*)carve((size_t)D_ * D_ * 2);
  __bf16* wkb  = (__bf16*)carve((size_t)D_ * D_ * 2);
  __bf16* wvb  = (__bf16*)carve((size_t)D_ * D_ * 2);
  __bf16* wpb  = (__bf16*)carve((size_t)D_ * D_ * 2);
  int*    gidx = (int*)carve((size_t)B_ * 64 * 4);

  topk_kernel<<<B_, 256, 0, stream>>>(acc, gidx);

  const int nx = MTOT * D_;
  cvt_kernel<<<(nx / 4 + 255) / 256, 256, 0, stream>>>(x, xb, nx);
  const int nw = D_ * D_;
  const int wblk = (nw / 4 + 255) / 256;
  cvt_kernel<<<wblk, 256, 0, stream>>>(qw, wqb, nw);
  cvt_kernel<<<wblk, 256, 0, stream>>>(kw, wkb, nw);
  cvt_kernel<<<wblk, 256, 0, stream>>>(vw, wvb, nw);
  cvt_kernel<<<wblk, 256, 0, stream>>>(pw, wpb, nw);

  const int mtiles = (MTOT + 127) / 128;  // 289
  dim3 blk(128);
  // K = xb @ kw^T (row-major bf16)
  gemm_kernel<<<dim3(mtiles, 6), blk, 0, stream>>>(xb, wkb, nullptr, MTOT, 0, kbb, nullptr, nullptr);
  // V^T = (xb @ vw^T)^T (token-contiguous per feature)
  gemm_kernel<<<dim3(mtiles, 6), blk, 0, stream>>>(xb, wvb, nullptr, MTOT, 1, vtb, nullptr, nullptr);
  // Q only at selected tokens (gathered rows)
  gemm_kernel<<<dim3(MQ / 128, 6), blk, 0, stream>>>(xb, wqb, gidx, MQ, 0, qsel, nullptr, nullptr);
  // local attention, scatter merged rows into xb
  attn_kernel<<<dim3(B_, H_), blk, 0, stream>>>(qsel, kbb, vtb, gidx, xb);
  // out = merged @ pw^T + pb (f32)
  gemm_kernel<<<dim3(mtiles, 6), blk, 0, stream>>>(xb, wpb, nullptr, MTOT, 2, nullptr, out, pb);
}
